// LSTM_LongitudinalModeling_72868415144153
// MI455X (gfx1250) — matrix operand order repaired and checked by reading, not compiled, not run on hardware
//
#include <hip/hip_runtime.h>

// ---------------------------------------------------------------------------
// LSTM (145 -> 150 -> 150 -> 32), T=64, B=4096.
// Strategy: per-layer persistent workgroups over the time loop.
//   - Each WG owns 64 batch rows; h/c live in LDS across all 64 timesteps.
//   - Gates = [x_t ; h_{t-1}] x [W_ih^T ; W_hh^T] via v_wmma_f32_16x16x32_bf16
//     (bf16 operands, fp32 accumulate), weights pre-swizzled into a contiguous-k
//     CDNA5 WMMA B-operand lane layout (one 32B contiguous read per lane).
//   - Elementwise gate math in fp32 from LDS, h written back to LDS (bf16)
//     and to a padded global bf16 buffer feeding the next layer.
// ---------------------------------------------------------------------------

typedef __bf16 bf16;
typedef __attribute__((ext_vector_type(8)))  bf16  v8bf;
typedef __attribute__((ext_vector_type(16))) bf16  v16bf;
typedef __attribute__((ext_vector_type(8)))  float v8f;

#define T_STEPS 64
#define BSZ     4096
#define BM      64      // batch rows per workgroup
#define NTHREADS 128    // 4 waves of 32

// ---- A-operand loader (16-bit A 16x32 per ISA 7.12.2) ----------------------
// Lane L holds row (L&15); half g = L>>4 selects K subsets:
//   elements j=0..7  -> K = K0 + 16g + j        (16 contiguous bytes)
//   elements j=8..15 -> K = K0 + 16g + j        (16 contiguous bytes)
__device__ __forceinline__ v16bf load_a_chunk(const bf16* __restrict__ row, int K0, int g) {
    const v8bf* p0 = reinterpret_cast<const v8bf*>(row + K0 + 16 * g);
    const v8bf* p1 = reinterpret_cast<const v8bf*>(row + K0 + 16 * g + 8);
    v8bf lo = *p0;
    v8bf hi = *p1;
    v16bf a;
#pragma unroll
    for (int i = 0; i < 8; ++i) { a[i] = lo[i]; a[i + 8] = hi[i]; }
    return a;
}

// ---- B-operand loader: weights pre-swizzled so lane L reads 32 contiguous B
__device__ __forceinline__ v16bf load_b_block(const bf16* __restrict__ blk, int L) {
    const v8bf* p = reinterpret_cast<const v8bf*>(blk + L * 16);
    v8bf lo = p[0];
    v8bf hi = p[1];
    v16bf b;
#pragma unroll
    for (int i = 0; i < 8; ++i) { b[i] = lo[i]; b[i + 8] = hi[i]; }
    return b;
}

__device__ __forceinline__ float sigmoidf_fast(float x) {
    return 1.0f / (1.0f + __expf(-x));
}

// ---------------------------------------------------------------------------
// Prep: fp32 (rows, kin) -> bf16 (rows, kp) zero-padded row-major.
// ---------------------------------------------------------------------------
__global__ void convert_pad_kernel(const float* __restrict__ in, bf16* __restrict__ out,
                                   long rows, int kin, int kp) {
    long n = rows * (long)kp;
    for (long i = blockIdx.x * (long)blockDim.x + threadIdx.x; i < n;
         i += (long)gridDim.x * blockDim.x) {
        long r = i / kp;
        int  k = (int)(i - r * kp);
        out[i] = (k < kin) ? (bf16)in[r * kin + k] : (bf16)0.0f;
    }
}

// ---------------------------------------------------------------------------
// Prep: swizzle fp32 W (fourH x kin, row-major) into WMMA B-operand layout.
// B[k][n] = W[n][k].  Block (nt, kc) -> 512 bf16: lane L element j holds
//   n = nt*16 + (L&15),  k = kc*32 + 16*(L>>4) + j
// stored contiguously at out[(nt*KC + kc)*512 + L*16 + j].
// ---------------------------------------------------------------------------
__global__ void swizzle_b_kernel(const float* __restrict__ w, bf16* __restrict__ out,
                                 int fourH, int kin, int KP, int NP) {
    int KC = KP / 32;
    long total = (long)KP * NP;
    for (long i = blockIdx.x * (long)blockDim.x + threadIdx.x; i < total;
         i += (long)gridDim.x * blockDim.x) {
        long blk = i >> 9;
        int  rem = (int)(i & 511);
        int  L = rem >> 4, j = rem & 15;
        int  nt = (int)(blk / KC);
        int  kc = (int)(blk - (long)nt * KC);
        int  n = nt * 16 + (L & 15);
        int  k = kc * 32 + ((L >> 4) << 4) + j;
        float v = (n < fourH && k < kin) ? w[(long)n * kin + k] : 0.0f;
        out[i] = (bf16)v;
    }
}

// ---------------------------------------------------------------------------
// Fused LSTM layer: one kernel runs all T timesteps for a 64-row batch tile.
//   KINP: padded input dim (mult of 32)   HP: padded hidden dim (mult of 32)
//   NPAD: padded 4H (mult of 16)          H : true hidden dim
// Dynamic LDS: gates[BM*NPAD] f32 | c[BM*HP] f32 | bias[NPAD] f32 | h[BM*HP] bf16
// ---------------------------------------------------------------------------
template <int KINP, int HP, int NPAD, int H, bool FINAL>
__global__ void lstm_layer_kernel(const bf16* __restrict__ xin,        // (T*B, KINP) bf16
                                  const bf16* __restrict__ w_ih_sw,    // swizzled (KINP x NPAD)
                                  const bf16* __restrict__ w_hh_sw,    // swizzled (HP x NPAD)
                                  const float* __restrict__ b_ih,
                                  const float* __restrict__ b_hh,
                                  bf16* __restrict__ hseq_out,         // (T*B, HP) bf16 (if !FINAL)
                                  float* __restrict__ final_out) {     // (T*B, H) f32 (if FINAL)
    constexpr int KC_IN = KINP / 32;
    constexpr int KC_HH = HP / 32;
    constexpr int NT    = NPAD / 16;

    extern __shared__ char smem[];
    float* gates = reinterpret_cast<float*>(smem);      // BM*NPAD
    float* cst   = gates + BM * NPAD;                   // BM*HP
    float* bias  = cst + BM * HP;                       // NPAD
    bf16*  hbuf  = reinterpret_cast<bf16*>(bias + NPAD);// BM*HP

    const int tid = threadIdx.x;
    const int wv  = tid >> 5;        // wave id 0..3 -> M-tile
    const int L   = tid & 31;        // lane
    const int g   = L >> 4;
    const int lr  = L & 15;
    const int b0  = blockIdx.x * BM;

    // init: zero h (incl. padding cols, kept zero forever), zero c, stage bias
    for (int i = tid; i < BM * HP; i += NTHREADS) { cst[i] = 0.0f; hbuf[i] = (bf16)0.0f; }
    for (int n = tid; n < NPAD; n += NTHREADS)
        bias[n] = (n < 4 * H) ? (b_ih[n] + b_hh[n]) : 0.0f;
    __syncthreads();

    for (int t = 0; t < T_STEPS; ++t) {
        // ---- GEMM phase: gates(16x608 per wave) = x_t*Wih^T + h*Whh^T ------
        const bf16* xrow = xin + ((long)t * BSZ + b0 + 16 * wv + lr) * KINP;
        const bf16* hrow = hbuf + (16 * wv + lr) * HP;

        v16bf ax[KC_IN];
        v16bf ah[KC_HH];
#pragma unroll
        for (int kc = 0; kc < KC_IN; ++kc) ax[kc] = load_a_chunk(xrow, 32 * kc, g);
#pragma unroll
        for (int kc = 0; kc < KC_HH; ++kc) ah[kc] = load_a_chunk(hrow, 32 * kc, g);

#pragma unroll 1
        for (int nt = 0; nt < NT; ++nt) {
            v8f acc;
#pragma unroll
            for (int i = 0; i < 8; ++i) acc[i] = 0.0f;
#pragma unroll
            for (int kc = 0; kc < KC_IN; ++kc) {
                v16bf b = load_b_block(w_ih_sw + ((long)(nt * KC_IN + kc) << 9), L);
                acc = __builtin_amdgcn_wmma_f32_16x16x32_bf16(false, ax[kc], false, b,
                                                              (short)0, acc, false, false);
            }
#pragma unroll
            for (int kc = 0; kc < KC_HH; ++kc) {
                v16bf b = load_b_block(w_hh_sw + ((long)(nt * KC_HH + kc) << 9), L);
                acc = __builtin_amdgcn_wmma_f32_16x16x32_bf16(false, ah[kc], false, b,
                                                              (short)0, acc, false, false);
            }
            // C/D layout: lane L col = nt*16 + (L&15); VGPR v row = v + 8*(L>>4)
#pragma unroll
            for (int v = 0; v < 8; ++v)
                gates[(16 * wv + v + 8 * g) * NPAD + nt * 16 + lr] = acc[v];
        }
        __syncthreads();

        // ---- elementwise phase: i,f,g,o -> c,h -----------------------------
        for (int e = tid; e < BM * HP; e += NTHREADS) {
            int m = e / HP;
            int j = e - m * HP;
            float hval = 0.0f;
            if (j < H) {
                const float* gr = gates + (long)m * NPAD;
                float gi = gr[j]         + bias[j];
                float gf = gr[H + j]     + bias[H + j];
                float gg = gr[2 * H + j] + bias[2 * H + j];
                float go = gr[3 * H + j] + bias[3 * H + j];
                float ig = sigmoidf_fast(gi);
                float fg = sigmoidf_fast(gf);
                float gt = tanhf(gg);
                float og = sigmoidf_fast(go);
                float c  = fg * cst[e] + ig * gt;
                cst[e]   = c;
                hval     = og * tanhf(c);
            }
            hbuf[e] = (bf16)hval;
            if constexpr (FINAL) {
                if (j < H)
                    final_out[((long)t * BSZ + b0 + m) * H + j] = hval;
            } else {
                hseq_out[((long)t * BSZ + b0 + m) * HP + j] = (bf16)hval;
            }
        }
        __syncthreads();
    }
}

// ---------------------------------------------------------------------------
extern "C" void kernel_launch(void* const* d_in, const int* in_sizes, int n_in,
                              void* d_out, int out_size, void* d_ws, size_t ws_size,
                              hipStream_t stream) {
    (void)in_sizes; (void)n_in; (void)out_size;

    const float* x      = (const float*)d_in[0];
    const float* wih0   = (const float*)d_in[1];
    const float* whh0   = (const float*)d_in[2];
    const float* bih0   = (const float*)d_in[3];
    const float* bhh0   = (const float*)d_in[4];
    const float* wih1   = (const float*)d_in[5];
    const float* whh1   = (const float*)d_in[6];
    const float* bih1   = (const float*)d_in[7];
    const float* bhh1   = (const float*)d_in[8];
    const float* wih2   = (const float*)d_in[9];
    const float* whh2   = (const float*)d_in[10];
    const float* bih2   = (const float*)d_in[11];
    const float* bhh2   = (const float*)d_in[12];
    float* out = (float*)d_out;

    const long TB = (long)T_STEPS * BSZ;           // 262144 tokens

    // ---- workspace layout (bytes, all 16B aligned) ----
    char* ws = (char*)d_ws;
    const size_t SZ_SEQ = (size_t)TB * 160 * sizeof(bf16);   // 83,886,080
    const size_t SZ_W01 = (size_t)160 * 608 * sizeof(bf16);  // 194,560
    const size_t SZ_WI2 = (size_t)160 * 128 * sizeof(bf16);  // 40,960
    const size_t SZ_WH2 = (size_t)32 * 128 * sizeof(bf16);   // 8,192
    bf16* xin     = (bf16*)(ws);
    bf16* buf0    = (bf16*)(ws + SZ_SEQ);
    bf16* buf1    = (bf16*)(ws + 2 * SZ_SEQ);
    bf16* wih0_sw = (bf16*)(ws + 3 * SZ_SEQ);
    bf16* whh0_sw = (bf16*)(ws + 3 * SZ_SEQ + SZ_W01);
    bf16* wih1_sw = (bf16*)(ws + 3 * SZ_SEQ + 2 * SZ_W01);
    bf16* whh1_sw = (bf16*)(ws + 3 * SZ_SEQ + 3 * SZ_W01);
    bf16* wih2_sw = (bf16*)(ws + 3 * SZ_SEQ + 4 * SZ_W01);
    bf16* whh2_sw = (bf16*)(ws + 3 * SZ_SEQ + 4 * SZ_W01 + SZ_WI2);
    (void)ws_size;

    // ---- prep: convert + pad input, swizzle all weight matrices ----
    {
        long n = TB * 160;
        int blocks = (int)((n + 255) / 256);
        convert_pad_kernel<<<blocks, 256, 0, stream>>>(x, xin, TB, 145, 160);
    }
    {
        int blocks = (int)((160L * 608 + 255) / 256);
        swizzle_b_kernel<<<blocks, 256, 0, stream>>>(wih0, wih0_sw, 600, 145, 160, 608);
        swizzle_b_kernel<<<blocks, 256, 0, stream>>>(whh0, whh0_sw, 600, 150, 160, 608);
        swizzle_b_kernel<<<blocks, 256, 0, stream>>>(wih1, wih1_sw, 600, 150, 160, 608);
        swizzle_b_kernel<<<blocks, 256, 0, stream>>>(whh1, whh1_sw, 600, 150, 160, 608);
        swizzle_b_kernel<<<(int)((160L * 128 + 255) / 256), 256, 0, stream>>>(
            wih2, wih2_sw, 128, 150, 160, 128);
        swizzle_b_kernel<<<(int)((32L * 128 + 255) / 256), 256, 0, stream>>>(
            whh2, whh2_sw, 128, 32, 32, 128);
    }

    // ---- layer kernels: 64 WGs x 128 threads, persistent over T ----
    const int grid = BSZ / BM;  // 64
    // LDS: gates + c + bias + h
    const size_t smem01 = (size_t)BM * 608 * 4 + (size_t)BM * 160 * 4 + 608 * 4 + (size_t)BM * 160 * 2; // 219,520
    const size_t smem2  = (size_t)BM * 128 * 4 + (size_t)BM * 32 * 4 + 128 * 4 + (size_t)BM * 32 * 2;   // 45,568

    lstm_layer_kernel<160, 160, 608, 150, false>
        <<<grid, NTHREADS, smem01, stream>>>(xin, wih0_sw, whh0_sw, bih0, bhh0, buf0, nullptr);
    lstm_layer_kernel<160, 160, 608, 150, false>
        <<<grid, NTHREADS, smem01, stream>>>(buf0, wih1_sw, whh1_sw, bih1, bhh1, buf1, nullptr);
    lstm_layer_kernel<160, 32, 128, 32, true>
        <<<grid, NTHREADS, smem2, stream>>>(buf1, wih2_sw, whh2_sw, bih2, bhh2, nullptr, out);
}
